// CustomAttention_57784490001237
// MI455X (gfx1250) — matrix operand order repaired and checked by reading, not compiled, not run on hardware
//
#include <hip/hip_runtime.h>

typedef __attribute__((ext_vector_type(16))) _Float16 v16h;
typedef __attribute__((ext_vector_type(8)))  _Float16 v8h;
typedef __attribute__((ext_vector_type(8)))  float    v8f;

#define NB    4
#define NSEQ  4096
#define DH    64
#define BK    64            // keys per block step
#define KSTR  72            // K-tile row stride (halves), 144B, 16B aligned
#define VSTR  72            // Vt-tile row stride (halves), 144B, 16B aligned
#define QSCALE 0.015625f    // 4096^-0.5 (reference scales by seq_len^-0.5)

// ---------------------------------------------------------------------------
// DPP16 row_xmask butterfly: lane_in_row ^= XM, pure VALU (no LDS, no waits).
// Reduction domain = 16-lane row, matching the WMMA C-layout half-waves.
// ---------------------------------------------------------------------------
template <int XM>
__device__ __forceinline__ float dpp_xor(float x) {
    int xi = __builtin_bit_cast(int, x);
    int yi = __builtin_amdgcn_update_dpp(0, xi, 0x160 | XM, 0xf, 0xf, true);
    return __builtin_bit_cast(float, yi);
}

__device__ __forceinline__ float row16_max(float x) {
    x = fmaxf(x, dpp_xor<1>(x));
    x = fmaxf(x, dpp_xor<2>(x));
    x = fmaxf(x, dpp_xor<4>(x));
    x = fmaxf(x, dpp_xor<8>(x));
    return x;
}

__device__ __forceinline__ float row16_sum(float x) {
    x += dpp_xor<1>(x);
    x += dpp_xor<2>(x);
    x += dpp_xor<4>(x);
    x += dpp_xor<8>(x);
    return x;
}

// ---------------------------------------------------------------------------
// QKV projection: y = x @ W.T, emit fp16. Scale folded into Q.
// grid: (B*N)/4 blocks of 256 threads; thread = (row-in-block r, out-col o)
// ---------------------------------------------------------------------------
__global__ __launch_bounds__(256) void qkv_proj(
    const float* __restrict__ x1, const float* __restrict__ x2,
    const float* __restrict__ Wq, const float* __restrict__ Wk,
    const float* __restrict__ Wv,
    _Float16* __restrict__ q, _Float16* __restrict__ k, _Float16* __restrict__ v)
{
    __shared__ float sx1[4][DH];
    __shared__ float sx2[4][DH];
    const int tid = threadIdx.x;
    const int r = tid >> 6;
    const int o = tid & 63;
    const size_t row = (size_t)blockIdx.x * 4 + r;
    sx1[r][o] = x1[row * DH + o];
    sx2[r][o] = x2[row * DH + o];
    __syncthreads();

    const float* wq = Wq + o * DH;   // W row-major [out][in]; y = x @ W.T
    const float* wk = Wk + o * DH;
    const float* wv = Wv + o * DH;
    float aq = 0.f, ak = 0.f, av = 0.f;
#pragma unroll
    for (int d = 0; d < DH; ++d) {
        aq = fmaf(sx1[r][d], wq[d], aq);
        ak = fmaf(sx2[r][d], wk[d], ak);
        av = fmaf(sx2[r][d], wv[d], av);
    }
    q[row * DH + o] = (_Float16)(aq * QSCALE);
    k[row * DH + o] = (_Float16)ak;
    v[row * DH + o] = (_Float16)av;
}

// ---------------------------------------------------------------------------
// Flash attention, one 16-row Q tile per wave, 4 waves (64 rows) per block.
// 64 keys per iteration: 16 WMMAs per softmax/barrier pass.
// ---------------------------------------------------------------------------
__global__ __launch_bounds__(128) void flash_attn(
    const _Float16* __restrict__ q, const _Float16* __restrict__ k,
    const _Float16* __restrict__ v, float* __restrict__ out)
{
    __shared__ _Float16 sK[BK * KSTR];        // [key][d], row-major, padded
    __shared__ _Float16 sV[DH * VSTR];        // [d][key], transposed, padded
    __shared__ _Float16 sP[4 * 16 * BK];      // per-wave P scratch (16x64)

    const int tid  = threadIdx.x;
    const int wave = tid >> 5;
    const int lane = tid & 31;
    const int l16  = lane & 15;
    const int hi   = lane >> 4;               // half-wave select

    const int blk  = blockIdx.x;              // 0..255
    const int b    = blk >> 6;                // batch
    const int row0 = (blk & 63) * 64 + wave * 16;

    const _Float16* qb = q + ((size_t)b * NSEQ + row0) * DH;
    const _Float16* kb = k + (size_t)b * NSEQ * DH;
    const _Float16* vb = v + (size_t)b * NSEQ * DH;

    // --- Q A-fragments (16x32 each; D=64 -> two k-chunks), ISA A layout:
    // lanes<16: row=l16, K={c*32+0..7, c*32+16..23}; lanes>=16: K shifted +8.
    v16h qa[2];
#pragma unroll
    for (int c = 0; c < 2; ++c) {
        const _Float16* p = qb + l16 * DH + c * 32 + hi * 8;
        v8h lo = *(const v8h*)p;
        v8h h8 = *(const v8h*)(p + 16);
#pragma unroll
        for (int i = 0; i < 8; ++i) { qa[c][i] = lo[i]; qa[c][i + 8] = h8[i]; }
    }

    // --- online-softmax state (C layout: VGPR r holds row r / r+8 per half)
    v8f acc[4];
#pragma unroll
    for (int n = 0; n < 4; ++n)
#pragma unroll
        for (int e = 0; e < 8; ++e) acc[n][e] = 0.0f;
    float m[8], l[8];
#pragma unroll
    for (int r = 0; r < 8; ++r) { m[r] = -1e30f; l[r] = 0.0f; }

    for (int kb0 = 0; kb0 < NSEQ; kb0 += BK) {
        __syncthreads();
        // --- cooperative stage: K row-major, V transposed (d-major)
        {
            const int kr = tid >> 1;            // 0..63 key row
            const int dc = (tid & 1) << 5;      // 0 or 32  d chunk
            const _Float16* gk = kb + (size_t)(kb0 + kr) * DH + dc;
#pragma unroll
            for (int j = 0; j < 4; ++j)
                *(v8h*)&sK[kr * KSTR + dc + j * 8] = *(const v8h*)(gk + j * 8);
            const _Float16* gv = vb + (size_t)(kb0 + kr) * DH + dc;
#pragma unroll
            for (int jj = 0; jj < 4; ++jj) {
                v8h vv = *(const v8h*)(gv + jj * 8);
#pragma unroll
                for (int j = 0; j < 8; ++j)
                    sV[(dc + jj * 8 + j) * VSTR + kr] = vv[j];
            }
            if (kb0 + BK < NSEQ) {              // prefetch next tiles into L2
                __builtin_prefetch(gk + BK * DH, 0, 0);
                __builtin_prefetch(gv + BK * DH, 0, 0);
            }
        }
        __syncthreads();

        // --- S = Q*K^T : four 16x16 key sub-tiles, K-dim 64 = 2 WMMA each
        v8f s[4];
#pragma unroll
        for (int t = 0; t < 4; ++t) {
            v8f c;
#pragma unroll
            for (int e = 0; e < 8; ++e) c[e] = 0.0f;
#pragma unroll
            for (int c2 = 0; c2 < 2; ++c2) {
                // B frag: lanes<16 hold K^=d c2*32+{0..7,16..23} for key n=l16;
                // lanes>=16 hold d shifted +8 (matches A layout)
                const _Float16* p = &sK[(t * 16 + l16) * KSTR + c2 * 32 + hi * 8];
                v8h b0 = *(const v8h*)p;
                v8h b1 = *(const v8h*)(p + 16);
                v16h bf;
#pragma unroll
                for (int i = 0; i < 8; ++i) { bf[i] = b0[i]; bf[i + 8] = b1[i]; }
                c = __builtin_amdgcn_wmma_f32_16x16x32_f16(
                        false, qa[c2], false, bf, (short)0, c, false, false);
            }
            s[t] = c;
        }

        // --- online softmax; row reductions are DPP row_xmask butterflies
        _Float16* myP = &sP[wave * 16 * BK];
#pragma unroll
        for (int r = 0; r < 8; ++r) {
            const float mx   = row16_max(fmaxf(fmaxf(s[0][r], s[1][r]),
                                               fmaxf(s[2][r], s[3][r])));
            const float mnew = fmaxf(m[r], mx);
            const float corr = __expf(m[r] - mnew);
            m[r] = mnew;
            float p[4];
#pragma unroll
            for (int t = 0; t < 4; ++t) p[t] = __expf(s[t][r] - mnew);
            const float rs = row16_sum((p[0] + p[1]) + (p[2] + p[3]));
            l[r] = l[r] * corr + rs;
#pragma unroll
            for (int n = 0; n < 4; ++n) acc[n][r] *= corr;
            // C-layout -> row-major P scratch (row = r + hi*8, col = t*16+l16)
#pragma unroll
            for (int t = 0; t < 4; ++t)
                myP[(r + hi * 8) * BK + t * 16 + l16] = (_Float16)p[t];
        }
        asm volatile("s_wait_dscnt 0" ::: "memory");

        // --- P as two A-fragments (16x32 each over the 64 keys)
        v16h pa[2];
#pragma unroll
        for (int c2 = 0; c2 < 2; ++c2) {
            const _Float16* pp = myP + l16 * BK + c2 * 32 + hi * 8;
            v8h lo = *(const v8h*)pp;
            v8h h8 = *(const v8h*)(pp + 16);
#pragma unroll
            for (int i = 0; i < 8; ++i) { pa[c2][i] = lo[i]; pa[c2][i + 8] = h8[i]; }
        }

        // --- O += P * V : 2 key-chunks x 4 output column tiles = 8 WMMAs
#pragma unroll
        for (int n = 0; n < 4; ++n) {
#pragma unroll
            for (int c2 = 0; c2 < 2; ++c2) {
                // B frag: lanes<16: col=16n+l16, keys c2*32+{0..7,16..23}; lanes>=16 +8
                const _Float16* p = &sV[(n * 16 + l16) * VSTR + c2 * 32 + hi * 8];
                v8h b0 = *(const v8h*)p;
                v8h b1 = *(const v8h*)(p + 16);
                v16h bf;
#pragma unroll
                for (int i = 0; i < 8; ++i) { bf[i] = b0[i]; bf[i + 8] = b1[i]; }
                acc[n] = __builtin_amdgcn_wmma_f32_16x16x32_f16(
                             false, pa[c2], false, bf, (short)0, acc[n], false, false);
            }
        }
    }

    // --- normalize and write (fp32 out; row = row0 + r + hi*8, col = n*16+l16)
#pragma unroll
    for (int r = 0; r < 8; ++r) {
        const float inv = 1.0f / l[r];
        const int row = row0 + r + hi * 8;
        float* po = out + ((size_t)b * NSEQ + row) * DH + l16;
#pragma unroll
        for (int n = 0; n < 4; ++n) po[n * 16] = acc[n][r] * inv;
    }
}

// ---------------------------------------------------------------------------
extern "C" void kernel_launch(void* const* d_in, const int* in_sizes, int n_in,
                              void* d_out, int out_size, void* d_ws, size_t ws_size,
                              hipStream_t stream) {
    (void)in_sizes; (void)n_in; (void)out_size; (void)ws_size;
    const float* x1 = (const float*)d_in[0];
    const float* x2 = (const float*)d_in[1];
    const float* Wq = (const float*)d_in[2];
    const float* Wk = (const float*)d_in[3];
    const float* Wv = (const float*)d_in[4];
    float* out = (float*)d_out;

    const size_t nElem = (size_t)NB * NSEQ * DH;    // 1 Mi elements
    _Float16* qh = (_Float16*)d_ws;                 // 2 MB
    _Float16* kh = qh + nElem;                      // 2 MB
    _Float16* vh = kh + nElem;                      // 2 MB

    qkv_proj<<<(NB * NSEQ) / 4, 256, 0, stream>>>(x1, x2, Wq, Wk, Wv, qh, kh, vh);
    flash_attn<<<NB * (NSEQ / 64), 128, 0, stream>>>(qh, kh, vh, out);
}
